// TrafficNet_61770219651522
// MI455X (gfx1250) — matrix operand order repaired and checked by reading, not compiled, not run on hardware
//
#include <hip/hip_runtime.h>

typedef __attribute__((ext_vector_type(16))) _Float16 v16h;
typedef __attribute__((ext_vector_type(8)))  float    v8f;

#define BETA    0.95f
#define THRESH  1.0f
#define NSTEPS  20
#define NH      64
#define NO      2
#define NI      2

// ---------------------------------------------------------------------------
// Prep kernel (32 threads, runs once per launch): split fc2 weights
// w = f16(w) + f16(w - f16(w)) and pack them into the exact 32x16 f16
// WMMA-B-matrix lane layout.  Table: [chunk c][hi/lo][lane][8 x f16-pair]
// = 4 KB, written as packed uint32 pairs.
// ---------------------------------------------------------------------------
__global__ void snn_prep(const float* __restrict__ fc2_w,
                         unsigned int* __restrict__ wtab32) {
    const int lane = threadIdx.x;      // 0..31
    const int m    = lane & 15;        // B-matrix column
    const int hiL  = lane >> 4;
    const bool active = (m < NO);
    for (int c = 0; c < 2; ++c) {
        for (int p = 0; p < 8; ++p) {  // f16 pair index (elements 2p, 2p+1)
            union { _Float16 h[2]; unsigned int u; } phi, plo;
            for (int j = 0; j < 2; ++j) {
                const int kk = c * 32 + (hiL ? 8 : 0) + ((p >= 4) ? 16 : 0) + (p & 3) * 2 + j;
                float w = active ? fc2_w[m * NH + kk] : 0.0f;
                _Float16 hi = (_Float16)w;
                phi.h[j] = hi;
                plo.h[j] = (_Float16)(w - (float)hi);
            }
            wtab32[((c * 2 + 0) * 32 + lane) * 8 + p] = phi.u;
            wtab32[((c * 2 + 1) * 32 + lane) * 8 + p] = plo.u;
        }
    }
}

// ---------------------------------------------------------------------------
// Main kernel: one wave = 16 batch rows.  mem1/cur1 sit in registers in the
// exact V_WMMA_F32_16X16X32_F16 A-matrix layout; the persistent spike matrix
// spkh (packed f16, 0/1) IS the WMMA A operand and also supplies the reset
// term of the next step (r1[s] == spk1[s-1]).  cur2 = spk1 @ W2^T via 4
// chained f16 WMMAs (split-f16 weights -> f32-exact since A is {0,1}).
// cur2 is transposed through wave-private LDS (same-wave in-order DS, no
// block barriers); the output LIF runs one element per lane and streams out
// as coalesced non-temporal b32 stores through running pointers.  Loop
// rotation lif1(s) -> emit(s-1) -> wmma(s) keeps ~130 independent VALU ops
// between the WMMA chain and the first read of its D registers.
// __launch_bounds__(256,1): one 8-wave block per WGP already means 2
// waves/SIMD; give the register allocator the full file so the four B-matrix
// v16h operands stay resident instead of being re-loaded in-loop.
// ---------------------------------------------------------------------------
__global__ __launch_bounds__(256, 1) void snn_lif_wmma(
    const float* __restrict__ x,
    const float* __restrict__ fc1_w,    // [NH, NI]
    const float* __restrict__ fc1_b,    // [NH]
    const _Float16* __restrict__ wtab,  // packed split-f16 B-matrix table
    const float* __restrict__ fc2_b,    // [NO]
    float* __restrict__ spk_out,        // [NSTEPS, B, NO]
    float* __restrict__ mem_out,        // [NSTEPS, B, NO]
    int B)
{
    __shared__ float lds[8][32];       // wave-private cur2 transpose tile

    const int tid  = threadIdx.x;
    const int wave = tid >> 5;
    const int lane = tid & 31;
    const int m    = lane & 15;        // row within tile
    const int hiL  = lane >> 4;

    const int rowBase = (int)blockIdx.x * 128 + wave * 16;
    const int rowG    = rowBase + m;

    // ---- B operands: four 32-byte vector loads, loop-invariant -------------
    const v16h* wt = (const v16h*)wtab;
    const v16h bHi0 = wt[0 * 32 + lane];
    const v16h bLo0 = wt[1 * 32 + lane];
    const v16h bHi1 = wt[2 * 32 + lane];
    const v16h bLo1 = wt[3 * 32 + lane];

    // ---- fc1 once: cur1 = x @ W1^T + b1, in WMMA-A layout ------------------
    const float x0 = x[(size_t)rowG * NI + 0];
    const float x1 = x[(size_t)rowG * NI + 1];

    float mem1[2][16];
    float cur1[2][16];
    v16h  spkh[2];                     // persistent spike matrix (f16 0/1)
    #pragma unroll
    for (int c = 0; c < 2; ++c) {
        #pragma unroll
        for (int e = 0; e < 16; ++e) {
            const int v = e >> 1, j = e & 1;
            const int k = c * 32 + (hiL ? 8 : 0) + ((v >= 4) ? 16 : 0)
                        + (v & 3) * 2 + j;
            cur1[c][e] = fmaf(x0, fc1_w[k * NI + 0],
                         fmaf(x1, fc1_w[k * NI + 1], fc1_b[k]));
            mem1[c][e] = 0.0f;
            spkh[c][e] = (_Float16)0.0f;
        }
    }

    // ---- transposed-side (per-lane) output-layer state ---------------------
    const bool  active = (m < NO);
    const int   cBase  = hiL * 16 + m;      // LDS write base (active lanes)
    const float biasT  = fc2_b[lane & 1];   // lane l -> (row l/2, n l%2)
    float mem2T = 0.0f;
    float spk2T = 0.0f;                     // doubles as reset term r2

    // Running output pointers: stride B*NO floats per step.
    float*       spkP   = spk_out + (size_t)rowBase * NO + lane;
    float*       memP   = mem_out + (size_t)rowBase * NO + lane;
    const size_t stride = (size_t)B * NO;

    // ---- layer-1 LIF update (prev spike = reset term, emits new spikes) ----
    auto lif1 = [&]() {
        #pragma unroll
        for (int c = 0; c < 2; ++c) {
            #pragma unroll
            for (int e = 0; e < 16; ++e) {
                // fma_mix: mn = (beta*mem + cur) - cvt_f32(spk[s-1])
                float mn = fmaf((float)spkh[c][e], -1.0f,
                                fmaf(BETA, mem1[c][e], cur1[c][e]));
                mem1[c][e] = mn;
                spkh[c][e] = (mn > THRESH) ? (_Float16)1.0f : (_Float16)0.0f;
            }
        }
    };

    // ---- cur2 = spk1 @ W2^T : 4 chained f16 WMMAs, f32 accumulate ----------
    auto gemv2 = [&]() -> v8f {
        v8f acc = {};
        acc = __builtin_amdgcn_wmma_f32_16x16x32_f16(false, spkh[0], false,
                  bHi0, (short)0, acc, false, false);
        acc = __builtin_amdgcn_wmma_f32_16x16x32_f16(false, spkh[0], false,
                  bLo0, (short)0, acc, false, false);
        acc = __builtin_amdgcn_wmma_f32_16x16x32_f16(false, spkh[1], false,
                  bHi1, (short)0, acc, false, false);
        acc = __builtin_amdgcn_wmma_f32_16x16x32_f16(false, spkh[1], false,
                  bLo1, (short)0, acc, false, false);
        return acc;
    };

    // ---- transpose cur2, output-layer LIF, streamed stores -----------------
    auto emit = [&](const v8f& acc) {
        if (active) {                        // single EXEC toggle per step
            #pragma unroll
            for (int r = 0; r < 8; ++r)      // C/D layout: VGPR r <-> row
                lds[wave][cBase + r * 2] = acc[r];
        }
        float cur2 = lds[wave][lane] + biasT;      // same-wave, in-order LDS
        float mn   = fmaf(BETA, mem2T, cur2) - spk2T;   // r2 == spk2[s-1]
        mem2T = mn;
        spk2T = (mn > THRESH) ? 1.0f : 0.0f;
        __builtin_nontemporal_store(spk2T, spkP);
        __builtin_nontemporal_store(mn,    memP);
        spkP += stride;
        memP += stride;
    };

    // ---- rotated, single-accumulator time loop -----------------------------
    lif1();                          // step 0 hidden update
    v8f acc = gemv2();               // step 0 matrix op
    #pragma unroll 1
    for (int s = 1; s < NSTEPS; ++s) {
        lif1();                      // step s hidden update (fills WMMA shadow)
        emit(acc);                   // step s-1 output path (first read of D)
        acc = gemv2();               // step s matrix op
    }
    emit(acc);                       // step NSTEPS-1 output path
}

extern "C" void kernel_launch(void* const* d_in, const int* in_sizes, int n_in,
                              void* d_out, int out_size, void* d_ws, size_t ws_size,
                              hipStream_t stream) {
    const float* x     = (const float*)d_in[0];
    const float* fc1_w = (const float*)d_in[1];
    const float* fc1_b = (const float*)d_in[2];
    const float* fc2_w = (const float*)d_in[3];
    const float* fc2_b = (const float*)d_in[4];

    const int B = in_sizes[0] / NI;              // 262144

    float* spk = (float*)d_out;                            // [NSTEPS, B, NO]
    float* mem = spk + (size_t)NSTEPS * (size_t)B * NO;    // [NSTEPS, B, NO]

    unsigned int* wtab32 = (unsigned int*)d_ws;  // 4 KB split-f16 weight table

    snn_prep<<<dim3(1), dim3(32), 0, stream>>>(fc2_w, wtab32);

    dim3 block(256);
    dim3 grid((unsigned)(B / 128));              // 8 waves x 16 rows per block
    snn_lif_wmma<<<grid, block, 0, stream>>>(x, fc1_w, fc1_b,
                                             (const _Float16*)wtab32, fc2_b,
                                             spk, mem, B);
}
